// Conv3DSA_37211596652588
// MI455X (gfx1250) — matrix operand order repaired and checked by reading, not compiled, not run on hardware
//
#include <hip/hip_runtime.h>

// ---------------------------------------------------------------------------
// Conv3D self-attention (Patt over HW=4096 per (c,d) slice, Datt over D=3).
// N=1, C=4, D=3, H=W=64.  HW=4096, per-channel count 12288, total 49152.
//
// E_ij = exp(q_i q_j) is SYMMETRIC, so row sums Z_i equal column sums.
// Both softmax passes reduce to:  out_j = sum_i w_i * exp(q_i * q_j)
//   pass 1: w = 1      -> Z          pass 2: w = v / Z  -> Patt
// Each 16x16 q_i*q_j tile is one V_WMMA_F32_16X16X32_F16 rank-1 outer product.
// Each wave drives TWO j-tiles off one shared A; tile consumption is software-
// pipelined one i-step behind the WMMA issue so the WMMA->VALU RAW hazard
// window is filled by the previous tile's exp/fma stream (no v_nop padding).
// q lives in LDS as zero-extended 32-bit halfwords so A/B are built from one
// aligned ds_load_b32 (A's K=1 garbage is killed by zero B row 1).
// q is pre-scaled by sqrt(log2 e); q >= 0 post-ReLU so tiles are >= 0 and raw
// v_exp_f32 (__builtin_amdgcn_exp2f) is exact -- no libm range fixup.
// Hi-lane A/B halves land in K=8 / K=16 slots whose partner rows are zero.
// ---------------------------------------------------------------------------

typedef __attribute__((ext_vector_type(16))) _Float16     v16h;
typedef __attribute__((ext_vector_type(8)))  float        v8f;
typedef __attribute__((ext_vector_type(8)))  unsigned int v8u;

#define CC 4
#define DD 3
#define HW 4096
#define PERC 12288          // D*H*W
#define TOT  49152          // C*D*H*W
#define SLICES 12           // C*D
#define SQRT_LOG2E 1.2011224087864498f   // sqrt(log2(e))

#if __has_builtin(__builtin_amdgcn_exp2f)
__device__ __forceinline__ float fast_exp2(float x) { return __builtin_amdgcn_exp2f(x); }
#else
__device__ __forceinline__ float fast_exp2(float x) { return __expf(x * 0.6931471805599453f); }
#endif

// ------------------------- conv3 + conv1 + partial stats --------------------
__global__ __launch_bounds__(256)
void conv_kernel(const float* __restrict__ x,
                 const float* __restrict__ W3, const float* __restrict__ b3,
                 const float* __restrict__ W1, const float* __restrict__ b1,
                 float* __restrict__ y3, float* __restrict__ y1,
                 float* __restrict__ partial) {
    __shared__ float  w3s[108];          // W3[c] : 4*27 taps (c uniform per block)
    __shared__ float4 red[256];
    const int tid = threadIdx.x;
    const int e   = blockIdx.x * 256 + tid;          // < TOT, PERC % 256 == 0
    const int c   = e / PERC;
    if (tid < 108) w3s[tid] = W3[c * 108 + tid];
    __syncthreads();

    const int rem = e % PERC;
    const int d   = rem >> 12;
    const int hw  = rem & 4095;
    const int h   = hw >> 6;
    const int w0  = hw & 63;

    // 3x3x3 conv, pad 1
    float a3 = b3[c];
#pragma unroll
    for (int ci = 0; ci < CC; ++ci) {
        const float* xc = x + ci * PERC;
        const float* wr = &w3s[ci * 27];
#pragma unroll
        for (int kd = 0; kd < 3; ++kd) {
            const int zd = d + kd - 1;
            if ((unsigned)zd >= (unsigned)DD) continue;
            const float* xd = xc + zd * HW;
#pragma unroll
            for (int kh = 0; kh < 3; ++kh) {
                const int zh = h + kh - 1;
                if ((unsigned)zh >= 64u) continue;
                const float* xr = xd + zh * 64;
#pragma unroll
                for (int kw = 0; kw < 3; ++kw) {
                    const int zw = w0 + kw - 1;
                    if ((unsigned)zw >= 64u) continue;
                    a3 = fmaf(wr[kd * 9 + kh * 3 + kw], xr[zw], a3);
                }
            }
        }
    }
    // 1x1x1 conv
    float a1 = b1[c];
#pragma unroll
    for (int ci = 0; ci < CC; ++ci)
        a1 = fmaf(W1[c * 4 + ci], x[ci * PERC + rem], a1);

    y3[e] = a3;
    y1[e] = a1;

    // deterministic per-block partial sums for BN batch stats
    red[tid] = make_float4(a3, a3 * a3, a1, a1 * a1);
    __syncthreads();
    for (int s = 128; s; s >>= 1) {
        if (tid < s) {
            float4 o = red[tid + s]; float4 m = red[tid];
            m.x += o.x; m.y += o.y; m.z += o.z; m.w += o.w;
            red[tid] = m;
        }
        __syncthreads();
    }
    if (tid == 0) {
        float4 m = red[0];
        float* p = partial + blockIdx.x * 4;
        p[0] = m.x; p[1] = m.y; p[2] = m.z; p[3] = m.w;
    }
}

// ---------------- finalize BN stats -> per-channel scale/shift --------------
__global__ void stats_kernel(const float* __restrict__ partial,
                             const float* __restrict__ g3, const float* __restrict__ be3,
                             const float* __restrict__ g1, const float* __restrict__ be1,
                             float* __restrict__ so) {
    const int c = threadIdx.x;
    if (c >= CC) return;
    float s0 = 0.f, s1 = 0.f, s2 = 0.f, s3 = 0.f;
    for (int b = 0; b < 48; ++b) {                    // 48 blocks per channel
        const float* p = partial + (c * 48 + b) * 4;
        s0 += p[0]; s1 += p[1]; s2 += p[2]; s3 += p[3];
    }
    const float inv = 1.0f / (float)PERC;
    const float m3 = s0 * inv, v3 = s1 * inv - m3 * m3;
    const float m1 = s2 * inv, v1 = s3 * inv - m1 * m1;
    const float sc3 = g3[c] * rsqrtf(v3 + 1e-5f);
    const float sc1 = g1[c] * rsqrtf(v1 + 1e-5f);
    so[c]      = sc3;  so[4 + c]  = be3[c] - m3 * sc3;
    so[8 + c]  = sc1;  so[12 + c] = be1[c] - m1 * sc1;
}

// ------------------------------ BN + ReLU -----------------------------------
__global__ __launch_bounds__(256)
void bnrelu_kernel(const float* __restrict__ y3, const float* __restrict__ y1,
                   const float* __restrict__ so,
                   float* __restrict__ v, float* __restrict__ kq) {
    const int e = blockIdx.x * 256 + threadIdx.x;
    const int c = e / PERC;
    const float a = fmaf(y3[e], so[c],     so[4 + c]);
    const float b = fmaf(y1[e], so[8 + c], so[12 + c]);
    v[e]  = a > 0.f ? a : 0.f;
    kq[e] = b > 0.f ? b : 0.f;
}

// ------------- WMMA weighted column-sum:  out_j = sum_i w_i exp(q_i q_j) ----
// grid = (16 j-chunks, 12 slices), block = 256 (8 waves, 32 columns per wave).
__global__ __launch_bounds__(256)
void attn_colsum_kernel(const float* __restrict__ q, const float* __restrict__ w,
                        float* __restrict__ out, int use_w) {
    __shared__ unsigned int qh32[HW + 32];         // f16 bits, zero-extended to u32
    __shared__ __align__(16) float wl[HW];         // 16 KB
    const int slice = blockIdx.y;
    const int tid   = threadIdx.x;
    const float* qs = q + slice * HW;
    for (int e = tid; e < HW; e += 256) {
        const _Float16 hv = (_Float16)(qs[e] * SQRT_LOG2E);   // fold exp->exp2 scale
        qh32[e] = (unsigned int)__builtin_bit_cast(unsigned short, hv);
        wl[e]   = use_w ? w[slice * HW + e] : 1.0f;
    }
    if (tid < 32) qh32[HW + tid] = 0u;             // zero pad for hi-lane reads
    __syncthreads();

    const int wave = tid >> 5;
    const int lane = tid & 31;
    const int j0   = blockIdx.x * 256 + wave * 32; // 32 columns per wave
    const int woff = (lane >> 4) << 3;             // lane's row-half offset for w

    // B tiles: lanes 0-15 -> K=0 row; hi lanes are zero (kills A's hi-lane K slot).
    v8u bu = {};
    bu[0] = (lane < 16) ? qh32[j0 + lane] : 0u;
    const v16h bv0 = __builtin_bit_cast(v16h, bu);
    bu[0] = (lane < 16) ? qh32[j0 + 16 + lane] : 0u;
    const v16h bv1 = __builtin_bit_cast(v16h, bu);

    float acc0[8], acc1[8];
#pragma unroll
    for (int r = 0; r < 8; ++r) { acc0[r] = 0.f; acc1[r] = 0.f; }
    const v8f cz = {};

    // ---- software pipeline: issue WMMAs for tile i, consume tile i-1 -------
    v8u au = {};
    au[0] = qh32[lane];                            // i0 = 0
    v8f p0 = __builtin_amdgcn_wmma_f32_16x16x32_f16(
        false, __builtin_bit_cast(v16h, au), false, bv0, (short)0, cz, false, false);
    v8f p1 = __builtin_amdgcn_wmma_f32_16x16x32_f16(
        false, __builtin_bit_cast(v16h, au), false, bv1, (short)0, cz, false, false);

    for (int i0 = 16; i0 < HW; i0 += 16) {
        au[0] = qh32[i0 + lane];                   // one ds_load_b32, shared A
        const v16h avv = __builtin_bit_cast(v16h, au);
        // next tiles issue first: previous tiles' exp/fma fill their hazard
        const v8f n0 = __builtin_amdgcn_wmma_f32_16x16x32_f16(
            false, avv, false, bv0, (short)0, cz, false, false);
        const v8f n1 = __builtin_amdgcn_wmma_f32_16x16x32_f16(
            false, avv, false, bv1, (short)0, cz, false, false);

        // consume previous tiles (rows i0-16 .. i0-1)
        const float4* wp = (const float4*)&wl[(i0 - 16) + woff];
        const float4 w01 = wp[0], w23 = wp[1];
        const float wv[8] = { w01.x, w01.y, w01.z, w01.w,
                              w23.x, w23.y, w23.z, w23.w };
#pragma unroll
        for (int r = 0; r < 8; ++r)
            acc0[r] = fmaf(wv[r], fast_exp2(p0[r]), acc0[r]);
#pragma unroll
        for (int r = 0; r < 8; ++r)
            acc1[r] = fmaf(wv[r], fast_exp2(p1[r]), acc1[r]);
        p0 = n0;
        p1 = n1;
    }
    {   // drain: consume last tiles (rows HW-16 .. HW-1)
        const float4* wp = (const float4*)&wl[(HW - 16) + woff];
        const float4 w01 = wp[0], w23 = wp[1];
        const float wv[8] = { w01.x, w01.y, w01.z, w01.w,
                              w23.x, w23.y, w23.z, w23.w };
#pragma unroll
        for (int r = 0; r < 8; ++r)
            acc0[r] = fmaf(wv[r], fast_exp2(p0[r]), acc0[r]);
#pragma unroll
        for (int r = 0; r < 8; ++r)
            acc1[r] = fmaf(wv[r], fast_exp2(p1[r]), acc1[r]);
    }

    float s0 = 0.f, s1 = 0.f;
#pragma unroll
    for (int r = 0; r < 8; ++r) { s0 += acc0[r]; s1 += acc1[r]; }
    s0 += __shfl_xor(s0, 16, 32);                  // fold rows 0-7 with 8-15
    s1 += __shfl_xor(s1, 16, 32);
    // lanes 0-15 own tile-0 columns, lanes 16-31 own tile-1 columns
    out[slice * HW + j0 + lane] = (lane < 16) ? s0 : s1;
}

// --------------------------- a = v / Z --------------------------------------
__global__ __launch_bounds__(256)
void avz_kernel(const float* __restrict__ v, const float* __restrict__ Z,
                float* __restrict__ a) {
    const int e = blockIdx.x * 256 + threadIdx.x;
    a[e] = v[e] / Z[e];
}

// -------------- depth attention (D=3) + fused epilogue ----------------------
__global__ __launch_bounds__(256)
void datt_final_kernel(const float* __restrict__ kq, const float* __restrict__ v,
                       const float* __restrict__ Patt, const float* __restrict__ x,
                       const float* __restrict__ gama, float* __restrict__ out) {
    const int t = blockIdx.x * 256 + threadIdx.x;   // C*HW = 16384
    if (t >= CC * HW) return;
    const int c  = t >> 12;
    const int hw = t & 4095;
    float q[DD], vv[DD], datt[DD];
#pragma unroll
    for (int i = 0; i < DD; ++i) {
        const int idx = (c * DD + i) * HW + hw;
        q[i] = kq[idx] * SQRT_LOG2E; vv[i] = v[idx]; datt[i] = 0.f;
    }
#pragma unroll
    for (int i = 0; i < DD; ++i) {
        float E[DD], z = 0.f;
#pragma unroll
        for (int j = 0; j < DD; ++j) { E[j] = fast_exp2(q[i] * q[j]); z += E[j]; }
        const float s = vv[i] / z;
#pragma unroll
        for (int j = 0; j < DD; ++j) datt[j] = fmaf(s, E[j], datt[j]);
    }
    const float g = gama[0];
#pragma unroll
    for (int j = 0; j < DD; ++j) {
        const int idx = (c * DD + j) * HW + hw;
        out[idx] = fmaf(g, Patt[idx] + datt[j], x[idx]);
    }
}

// ---------------------------------------------------------------------------
extern "C" void kernel_launch(void* const* d_in, const int* in_sizes, int n_in,
                              void* d_out, int out_size, void* d_ws, size_t ws_size,
                              hipStream_t stream) {
    const float* x    = (const float*)d_in[0];
    const float* W3   = (const float*)d_in[1];
    const float* b3   = (const float*)d_in[2];
    const float* g3   = (const float*)d_in[3];
    const float* be3  = (const float*)d_in[4];
    const float* W1   = (const float*)d_in[5];
    const float* b1   = (const float*)d_in[6];
    const float* g1   = (const float*)d_in[7];
    const float* be1  = (const float*)d_in[8];
    const float* gama = (const float*)d_in[9];
    float* out = (float*)d_out;

    float* ws      = (float*)d_ws;
    float* partial = ws;                 // 192 blocks * 4
    float* so      = ws + 768;           // 16 scale/shift values
    float* y3      = ws + 800;
    float* y1      = y3 + TOT;
    float* vb      = y1 + TOT;
    float* kq      = vb + TOT;
    float* Z       = kq + TOT;
    float* av      = Z  + TOT;
    float* Patt    = av + TOT;

    conv_kernel<<<192, 256, 0, stream>>>(x, W3, b3, W1, b1, y3, y1, partial);
    stats_kernel<<<1, 32, 0, stream>>>(partial, g3, be3, g1, be1, so);
    bnrelu_kernel<<<192, 256, 0, stream>>>(y3, y1, so, vb, kq);
    // pass 1: Z_i = sum_j exp(q_i q_j)  (symmetry: row sums == column sums)
    attn_colsum_kernel<<<dim3(16, SLICES), 256, 0, stream>>>(kq, Z, Z, 0);
    avz_kernel<<<192, 256, 0, stream>>>(vb, Z, av);
    // pass 2: Patt_j = sum_i (v_i/Z_i) exp(q_i q_j)
    attn_colsum_kernel<<<dim3(16, SLICES), 256, 0, stream>>>(kq, av, Patt, 1);
    datt_final_kernel<<<64, 256, 0, stream>>>(kq, vb, Patt, x, gama, out);
}
